// FlashSelfAttention_32083405701156
// MI455X (gfx1250) — matrix operand order repaired and checked by reading, compiled, not run on hardware
//
#include <hip/hip_runtime.h>

typedef __bf16 bf16_t;
typedef __attribute__((ext_vector_type(16))) __bf16 v16bf;
typedef __attribute__((ext_vector_type(8)))  __bf16 v8bf;
typedef __attribute__((ext_vector_type(8)))  float   v8f;

#define D_MODEL 512
#define NHEADS  8
#define HDIM    64
#define BATCH   2
#define SEQ     2048

#define TM 64
#define TN 64
#define TK 32
#define PAD 8

// ---------------------------------------------------------------------------
// CDNA5 wave32 WMMA fragment loaders (layouts per CDNA5 ISA §7.12.2)
// Every fragment is exactly 2x ds_load_b128 per lane.
// ---------------------------------------------------------------------------

__device__ __forceinline__ v16bf cat8(v8bf lo, v8bf hi) {
  return __builtin_shufflevector(lo, hi, 0, 1, 2, 3, 4, 5, 6, 7,
                                         8, 9, 10, 11, 12, 13, 14, 15);
}

// A-matrix 16x32 bf16 fragment from LDS tile stored row-major [m][k], stride LD.
template<int LD>
__device__ __forceinline__ v16bf load_a_frag(const bf16_t* base) {
  const int lane = threadIdx.x & 31;
  const bf16_t* p = base + (lane & 15) * LD + (lane >> 4) * 8;
  v8bf lo = *(const v8bf*)(p);        // K 0..7   (or 8..15)
  v8bf hi = *(const v8bf*)(p + 16);   // K 16..23 (or 24..31)
  return cat8(lo, hi);
}

// B-matrix 32x16 bf16 fragment from LDS tile stored n-major [n][k], stride LD.
template<int LD>
__device__ __forceinline__ v16bf load_b_frag_nmajor(const bf16_t* base) {
  const int lane = threadIdx.x & 31;
  const bf16_t* p = base + (lane & 15) * LD + (lane >> 4) * 8;
  v8bf lo = *(const v8bf*)(p);
  v8bf hi = *(const v8bf*)(p + 16);
  return cat8(lo, hi);
}

__device__ __forceinline__ v8f wmma_bf16(v16bf a, v16bf b, v8f c) {
  return __builtin_amdgcn_wmma_f32_16x16x32_bf16(false, a, false, b, (short)0, c,
                                                 false, false);
}

// CDNA5 async copy global->LDS (16B per lane), tracked by ASYNCcnt.
__device__ __forceinline__ void async_copy16(const bf16_t* gsrc, bf16_t* ldst) {
  unsigned laddr =
      (unsigned)(uintptr_t)(__attribute__((address_space(3))) bf16_t*)ldst;
  unsigned long long gaddr = (unsigned long long)(uintptr_t)gsrc;
  asm volatile("global_load_async_to_lds_b128 %0, %1, off"
               :: "v"(laddr), "v"(gaddr) : "memory");
}
__device__ __forceinline__ void async_wait0() {
  asm volatile("s_wait_asynccnt 0x0" ::: "memory");
}

// ---------------------------------------------------------------------------
// Pre-pass: one-time bf16 conversion of activations and (transposed) weights.
//   x  [4096,512] f32 -> Xb  [4096,512] bf16
//   W  [512,C]    f32 -> Wt  [C,512]    bf16   (n-major, GEMM-B ready)
// ---------------------------------------------------------------------------
__global__ __launch_bounds__(128) void cvt_copy_kernel(
    const float* __restrict__ src, bf16_t* __restrict__ dst) {
  size_t i = ((size_t)blockIdx.x * 128 + threadIdx.x) * 8;
  float4 a = ((const float4*)(src + i))[0];
  float4 b = ((const float4*)(src + i))[1];
  v8bf v;
  v[0] = (bf16_t)a.x; v[1] = (bf16_t)a.y; v[2] = (bf16_t)a.z; v[3] = (bf16_t)a.w;
  v[4] = (bf16_t)b.x; v[5] = (bf16_t)b.y; v[6] = (bf16_t)b.z; v[7] = (bf16_t)b.w;
  *(v8bf*)(dst + i) = v;
}

__global__ __launch_bounds__(128) void cvt_transpose_kernel(
    const float* __restrict__ src, bf16_t* __restrict__ dst, int C) {
  int gid = blockIdx.x * 128 + threadIdx.x;
  int c  = gid >> 6;             // output row (= W column), 64 threads per column
  int r0 = (gid & 63) * 8;       // k offset
  v8bf v;
#pragma unroll
  for (int j = 0; j < 8; ++j) v[j] = (bf16_t)src[(size_t)(r0 + j) * C + c];
  *(v8bf*)(dst + (size_t)c * D_MODEL + r0) = v;
}

// ---------------------------------------------------------------------------
// Kernel 1: fused QKV projection  qkv = Xb @ Wqt^T + bqkv  (bf16 WMMA, f32 acc)
//   writes Q/K bf16 in [B,H,N,HDIM]; V bf16 TRANSPOSED in [B,H,HDIM,N]
// ---------------------------------------------------------------------------
__global__ __launch_bounds__(128) void qkv_proj_kernel(
    const bf16_t* __restrict__ Ag, const bf16_t* __restrict__ Bt,
    const float* __restrict__ bqkv,
    bf16_t* __restrict__ Qb, bf16_t* __restrict__ Kb, bf16_t* __restrict__ Vb) {
  __shared__ bf16_t As[TM][TK + PAD];        // row-major [m][k]
  __shared__ bf16_t Bs[TN][TK + PAD];        // n-major  [n][k]

  const int row0 = blockIdx.x * TM;
  const int col0 = blockIdx.y * TN;
  const int tid  = threadIdx.x;
  const int w    = tid >> 5;
  const int lane = tid & 31;

  v8f zero8 = {0.f, 0.f, 0.f, 0.f, 0.f, 0.f, 0.f, 0.f};
  v8f acc[4];
#pragma unroll
  for (int f = 0; f < 4; ++f) acc[f] = zero8;

  for (int k0 = 0; k0 < D_MODEL; k0 += TK) {
    __syncthreads();
#pragma unroll
    for (int it = 0; it < 2; ++it) {
      int i = it * 128 + tid;
      int r = i >> 2, c4 = i & 3;
      async_copy16(Ag + (size_t)(row0 + r) * D_MODEL + k0 + c4 * 8, &As[r][c4 * 8]);
      async_copy16(Bt + (size_t)(col0 + r) * D_MODEL + k0 + c4 * 8, &Bs[r][c4 * 8]);
    }
    async_wait0();
    __syncthreads();

    v16bf a = load_a_frag<TK + PAD>(&As[w * 16][0]);
#pragma unroll
    for (int f = 0; f < 4; ++f) {
      v16bf b = load_b_frag_nmajor<TK + PAD>(&Bs[f * 16][0]);
      acc[f] = wmma_bf16(a, b, acc[f]);
    }
  }

  const int hi16 = lane >> 4;
  const int nloc = lane & 15;
#pragma unroll
  for (int f = 0; f < 4; ++f) {
#pragma unroll
    for (int r = 0; r < 8; ++r) {
      int m = row0 + w * 16 + r + hi16 * 8;   // flat token index (b*SEQ + n)
      int c = col0 + f * 16 + nloc;           // column in [0,1536)
      float val = acc[f][r] + bqkv[c];
      int bb  = m >> 11;
      int tok = m & (SEQ - 1);
      int t   = c >> 9;                       // 0=Q, 1=K, 2=V
      int h   = (c >> 6) & 7;
      int d   = c & 63;
      size_t bh = (size_t)(bb * NHEADS + h);
      if (t == 2) {
        Vb[(bh * HDIM + d) * SEQ + tok] = (bf16_t)val;        // V transposed
      } else {
        bf16_t* dst = (t == 0) ? Qb : Kb;
        dst[(bh * SEQ + tok) * HDIM + d] = (bf16_t)val;
      }
    }
  }
}

// ---------------------------------------------------------------------------
// Kernel 2: flash attention with bias + key padding mask
//   128 threads (4 wave32) own (b, h) x 64 query rows; K/V^T double-buffered
//   via async-to-LDS so tile i+1 loads overlap compute on tile i.
// ---------------------------------------------------------------------------
__global__ __launch_bounds__(128) void attn_kernel(
    const bf16_t* __restrict__ Qb, const bf16_t* __restrict__ Kb,
    const bf16_t* __restrict__ Vb, const float* __restrict__ bias,
    const unsigned char* __restrict__ kpm, bf16_t* __restrict__ Ob) {
  __shared__ bf16_t Qs[64][HDIM + PAD];         // [m][d]   A-frags
  __shared__ bf16_t Ks[2][32][HDIM + PAD];      // [key][d] n-major B-frags
  __shared__ bf16_t Vt[2][HDIM][32 + PAD];      // [d][key] n-major B-frags
  __shared__ bf16_t Ps[4][16][32 + PAD];        // per-wave P, A-frags

  const int tid  = threadIdx.x;
  const int w    = tid >> 5;
  const int lane = tid & 31;
  const int hi16 = lane >> 4;
  const int nloc = lane & 15;

  const int blk  = blockIdx.x;
  const int mblk = blk & 31;
  const int h    = (blk >> 5) & 7;
  const int b    = blk >> 8;
  const int m0   = mblk * 64;

  const float LOG2E  = 1.4426950408889634f;
  const float SSCALE = 0.125f * LOG2E;          // 1/sqrt(64) * log2(e)

  const size_t bh = (size_t)(b * NHEADS + h);
  const bf16_t* Qg = Qb + bh * SEQ * HDIM;
  const bf16_t* Kg = Kb + bh * SEQ * HDIM;
  const bf16_t* Vg = Vb + bh * (size_t)HDIM * SEQ;   // transposed [d][n]
  const float*  Bg = bias + bh * (size_t)SEQ * SEQ;

  // stage Q (async) and K/V tile 0
#pragma unroll
  for (int it = 0; it < 4; ++it) {
    int i = it * 128 + tid;
    int r = i >> 3, c8 = i & 7;
    async_copy16(Qg + (size_t)(m0 + r) * HDIM + c8 * 8, &Qs[r][c8 * 8]);
  }
#pragma unroll
  for (int it = 0; it < 2; ++it) {
    int i = it * 128 + tid;
    int kr = i >> 3, kc = i & 7;
    async_copy16(Kg + (size_t)kr * HDIM + kc * 8, &Ks[0][kr][kc * 8]);
    int vr = i >> 2, vc = i & 3;
    async_copy16(Vg + (size_t)vr * SEQ + vc * 8, &Vt[0][vr][vc * 8]);
  }
  async_wait0();
  __syncthreads();

  v16bf qa0 = load_a_frag<HDIM + PAD>(&Qs[w * 16][0]);
  v16bf qa1 = load_a_frag<HDIM + PAD>(&Qs[w * 16][32]);

  const int mrow_base = m0 + w * 16 + hi16 * 8;

  float rmax[8], rsum[8];
  v8f zero8 = {0.f, 0.f, 0.f, 0.f, 0.f, 0.f, 0.f, 0.f};
  v8f accO[4];
#pragma unroll
  for (int r = 0; r < 8; ++r) { rmax[r] = -3.0e38f; rsum[r] = 0.0f; }
#pragma unroll
  for (int f = 0; f < 4; ++f) accO[f] = zero8;

  const int NT = SEQ / 32;     // 64 key tiles
  for (int t = 0; t < NT; ++t) {
    const int n0  = t * 32;
    const int buf = t & 1;

    // issue async loads for tile t+1 into the other buffer (overlaps compute)
    if (t + 1 < NT) {
      const int n1 = n0 + 32;
      const int nb = buf ^ 1;
#pragma unroll
      for (int it = 0; it < 2; ++it) {
        int i = it * 128 + tid;
        int kr = i >> 3, kc = i & 7;
        async_copy16(Kg + (size_t)(n1 + kr) * HDIM + kc * 8, &Ks[nb][kr][kc * 8]);
        int vr = i >> 2, vc = i & 3;
        async_copy16(Vg + (size_t)vr * SEQ + n1 + vc * 8, &Vt[nb][vr][vc * 8]);
      }
      __builtin_prefetch(Bg + (size_t)mrow_base * SEQ + n1 + nloc, 0, 1);
    }

    // S = Q * K^T  (two 16-key tiles; wmma K-dim = head dim)
    v8f s[2];
#pragma unroll
    for (int nt = 0; nt < 2; ++nt) {
      v8f sacc = zero8;
      v16bf bk0 = load_b_frag_nmajor<HDIM + PAD>(&Ks[buf][nt * 16][0]);
      sacc = wmma_bf16(qa0, bk0, sacc);
      v16bf bk1 = load_b_frag_nmajor<HDIM + PAD>(&Ks[buf][nt * 16][32]);
      sacc = wmma_bf16(qa1, bk1, sacc);
      s[nt] = sacc;
    }

    // scale + bias (non-temporal 268MB stream) + key padding mask
#pragma unroll
    for (int nt = 0; nt < 2; ++nt) {
      int n = n0 + nt * 16 + nloc;
      float maskv = kpm[b * SEQ + n] ? 0.0f : -3.0e38f;
#pragma unroll
      for (int r = 0; r < 8; ++r) {
        float bvl = __builtin_nontemporal_load(Bg + (size_t)(mrow_base + r) * SEQ + n);
        s[nt][r] = fmaf(s[nt][r], SSCALE, fmaf(bvl, LOG2E, maskv));
      }
    }

    // online softmax (base-2); one row = 16 lanes of a half-wave
#pragma unroll
    for (int r = 0; r < 8; ++r) {
      float lmax = fmaxf(s[0][r], s[1][r]);
#pragma unroll
      for (int off = 8; off >= 1; off >>= 1)
        lmax = fmaxf(lmax, __shfl_xor(lmax, off, 16));
      float mnew  = fmaxf(rmax[r], lmax);
      float alpha = exp2f(rmax[r] - mnew);
      rmax[r] = mnew;
      float p0 = exp2f(s[0][r] - mnew);
      float p1 = exp2f(s[1][r] - mnew);
      float lsum = p0 + p1;
#pragma unroll
      for (int off = 8; off >= 1; off >>= 1)
        lsum += __shfl_xor(lsum, off, 16);
      rsum[r] = rsum[r] * alpha + lsum;
#pragma unroll
      for (int f = 0; f < 4; ++f) accO[f][r] = accO[f][r] * alpha;
      Ps[w][r + hi16 * 8][nloc]      = (bf16_t)p0;
      Ps[w][r + hi16 * 8][16 + nloc] = (bf16_t)p1;
    }

    // O += P * V
    v16bf pa = load_a_frag<32 + PAD>(&Ps[w][0][0]);
#pragma unroll
    for (int f = 0; f < 4; ++f) {
      v16bf vb = load_b_frag_nmajor<32 + PAD>(&Vt[buf][f * 16][0]);
      accO[f] = wmma_bf16(pa, vb, accO[f]);
    }

    // tile t+1 fully in LDS + everyone done reading buf before it is reused
    async_wait0();
    __syncthreads();
  }

  // normalize, write O bf16 in [B*N, D_MODEL]
#pragma unroll
  for (int r = 0; r < 8; ++r) {
    float inv = 1.0f / rsum[r];
    int tok = mrow_base + r;
#pragma unroll
    for (int f = 0; f < 4; ++f) {
      int c = h * HDIM + f * 16 + nloc;
      Ob[(size_t)(b * SEQ + tok) * D_MODEL + c] = (bf16_t)(accO[f][r] * inv);
    }
  }
}

// ---------------------------------------------------------------------------
// Kernel 3: output projection  out = Ob @ Wot^T + bout   (f32 result)
// ---------------------------------------------------------------------------
__global__ __launch_bounds__(128) void out_proj_kernel(
    const bf16_t* __restrict__ Ag, const bf16_t* __restrict__ Bt,
    const float* __restrict__ bout, float* __restrict__ Out) {
  __shared__ bf16_t As[TM][TK + PAD];
  __shared__ bf16_t Bs[TN][TK + PAD];

  const int row0 = blockIdx.x * TM;
  const int col0 = blockIdx.y * TN;
  const int tid  = threadIdx.x;
  const int w    = tid >> 5;
  const int lane = tid & 31;

  v8f zero8 = {0.f, 0.f, 0.f, 0.f, 0.f, 0.f, 0.f, 0.f};
  v8f acc[4];
#pragma unroll
  for (int f = 0; f < 4; ++f) acc[f] = zero8;

  for (int k0 = 0; k0 < D_MODEL; k0 += TK) {
    __syncthreads();
#pragma unroll
    for (int it = 0; it < 2; ++it) {
      int i = it * 128 + tid;
      int r = i >> 2, c4 = i & 3;
      async_copy16(Ag + (size_t)(row0 + r) * D_MODEL + k0 + c4 * 8, &As[r][c4 * 8]);
      async_copy16(Bt + (size_t)(col0 + r) * D_MODEL + k0 + c4 * 8, &Bs[r][c4 * 8]);
    }
    async_wait0();
    __syncthreads();

    v16bf a = load_a_frag<TK + PAD>(&As[w * 16][0]);
#pragma unroll
    for (int f = 0; f < 4; ++f) {
      v16bf b = load_b_frag_nmajor<TK + PAD>(&Bs[f * 16][0]);
      acc[f] = wmma_bf16(a, b, acc[f]);
    }
  }

  const int hi16 = lane >> 4;
  const int nloc = lane & 15;
#pragma unroll
  for (int f = 0; f < 4; ++f) {
#pragma unroll
    for (int r = 0; r < 8; ++r) {
      int m = row0 + w * 16 + r + hi16 * 8;
      int c = col0 + f * 16 + nloc;
      Out[(size_t)m * D_MODEL + c] = acc[f][r] + bout[c];
    }
  }
}

// ---------------------------------------------------------------------------
// launch
// ---------------------------------------------------------------------------
extern "C" void kernel_launch(void* const* d_in, const int* in_sizes, int n_in,
                              void* d_out, int out_size, void* d_ws, size_t ws_size,
                              hipStream_t stream) {
  (void)in_sizes; (void)n_in; (void)out_size; (void)ws_size;

  const float* x           = (const float*)d_in[0];
  const float* attn_bias   = (const float*)d_in[1];
  const unsigned char* kpm = (const unsigned char*)d_in[2];
  const float* Wqkv        = (const float*)d_in[3];
  const float* bqkv        = (const float*)d_in[4];
  const float* Wout        = (const float*)d_in[5];
  const float* bout        = (const float*)d_in[6];
  float* out               = (float*)d_out;

  const size_t per = (size_t)BATCH * NHEADS * SEQ * HDIM;   // 2Mi elems
  bf16_t* Qb  = (bf16_t*)d_ws;
  bf16_t* Kb  = Qb + per;
  bf16_t* Vb  = Kb + per;                       // transposed [B,H,HDIM,SEQ]
  bf16_t* Ob  = Vb + per;                       // [B*N, D_MODEL]
  bf16_t* Xb  = Ob + per;                       // [B*N, D_MODEL] bf16 copy of x
  bf16_t* Wqt = Xb + per;                       // [1536, 512] n-major
  bf16_t* Wot = Wqt + (size_t)(3 * D_MODEL) * D_MODEL;  // [512, 512] n-major

  // one-time bf16 conversion / weight transposition (tiny vs. 268MB bias)
  cvt_copy_kernel<<<dim3((BATCH * SEQ * D_MODEL) / 1024), 128, 0, stream>>>(x, Xb);
  cvt_transpose_kernel<<<dim3((3 * D_MODEL) / 2), 128, 0, stream>>>(Wqkv, Wqt,
                                                                    3 * D_MODEL);
  cvt_transpose_kernel<<<dim3(D_MODEL / 2), 128, 0, stream>>>(Wout, Wot, D_MODEL);

  qkv_proj_kernel<<<dim3((BATCH * SEQ) / TM, (3 * D_MODEL) / TN), 128, 0, stream>>>(
      Xb, Wqt, bqkv, Qb, Kb, Vb);

  attn_kernel<<<dim3(BATCH * NHEADS * (SEQ / 64)), 128, 0, stream>>>(
      Qb, Kb, Vb, attn_bias, kpm, Ob);

  out_proj_kernel<<<dim3((BATCH * SEQ) / TM, D_MODEL / TN), 128, 0, stream>>>(
      Ob, Wot, bout, out);
}
